// ButterflyProduct_7954279432739
// MI455X (gfx1250) — compile-verified, matrix operand order checked by reading, not run on hardware
//
#include <hip/hip_runtime.h>

// ---------------------------------------------------------------------------
// Butterfly product == product of 10 banded (21-diagonal) 1024x1024 operators.
// Plan: compose them into one dense T (workspace, L2-resident), then a single
// 16384x1024 @ 1024x1024 bf16 WMMA GEMM at the HBM roofline.
// ---------------------------------------------------------------------------

#define SIZE_N    1024
#define M_FACTORS 10
#define N_TERMS_C 10
#define BATCH_C   16384
#define NBANDS    21      // 1 diag + 10 sub + 10 super

typedef __attribute__((ext_vector_type(8)))  float  v8f;
typedef __attribute__((ext_vector_type(4)))  __bf16 v4bf;
typedef __attribute__((ext_vector_type(8)))  __bf16 v8bf;
typedef __attribute__((ext_vector_type(16))) __bf16 v16bf;

struct alignas(8) us4 { unsigned short x, y, z, w; };

// Hardware bf16 conversion (fptrunc -> v_cvt_*bf16_f32, RTNE)
__device__ __forceinline__ v4bf cvt4(float4 v) {
  v4bf o;
  o.x = (__bf16)v.x; o.y = (__bf16)v.y; o.z = (__bf16)v.z; o.w = (__bf16)v.w;
  return o;
}

#if __has_builtin(__builtin_amdgcn_global_load_async_to_lds_b128) && \
    __has_builtin(__builtin_amdgcn_s_wait_asynccnt)
#define USE_ASYNC_LDS 1
// Builtin signature (from the compiler's own diagnostic): arg0 is an AS1
// pointer to a 16-byte int vector, arg1 the AS3 (LDS) counterpart, then two
// immediates (offset, cpol).
typedef int v4i_g __attribute__((vector_size(16)));
typedef __attribute__((address_space(1))) v4i_g* as1_v4i;
typedef __attribute__((address_space(3))) v4i_g* as3_v4i;
#else
#define USE_ASYNC_LDS 0
#endif

// ---------------------------------------------------------------------------
// 1) Per-iteration softmax + combined band coefficients.
//    coef[i][b][j]: contribution to out[j] from x[j + off_b]
// ---------------------------------------------------------------------------
__global__ __launch_bounds__(256) void prep_bands(
    const float* __restrict__ logit, const float* __restrict__ diags,
    const float* __restrict__ sub,   const float* __restrict__ sup,
    float* __restrict__ coef) {
  const int i = blockIdx.x;
  float p[M_FACTORS];
  float mx = -3.402823e38f;
#pragma unroll
  for (int f = 0; f < M_FACTORS; ++f) { p[f] = logit[i * M_FACTORS + f]; mx = fmaxf(mx, p[f]); }
  float s = 0.f;
#pragma unroll
  for (int f = 0; f < M_FACTORS; ++f) { p[f] = __expf(p[f] - mx); s += p[f]; }
  const float inv = 1.f / s;
#pragma unroll
  for (int f = 0; f < M_FACTORS; ++f) p[f] *= inv;

  float* cb = coef + (size_t)i * NBANDS * SIZE_N;
  for (int j = threadIdx.x; j < SIZE_N; j += 256) {
    float cd = 0.f;
#pragma unroll
    for (int f = 0; f < M_FACTORS; ++f) cd += p[f] * diags[f * SIZE_N + j];
    cb[j] = cd;
#pragma unroll
    for (int f = 0; f < M_FACTORS; ++f) {
      const int d = 1 << (M_FACTORS - 1 - f);
      cb[(1 + f)  * SIZE_N + j] = (j >= d)         ? p[f] * sub[f * SIZE_N + (j - d)] : 0.f;
      cb[(11 + f) * SIZE_N + j] = (j < SIZE_N - d) ? p[f] * sup[f * SIZE_N + j]       : 0.f;
    }
  }
}

// ---------------------------------------------------------------------------
// 2) Materialize banded operator M_iter as dense 1024x1024 (f32).
// ---------------------------------------------------------------------------
__global__ __launch_bounds__(256) void band_init(const float* __restrict__ coef,
                                                 float* __restrict__ P, int iter) {
  const int j = blockIdx.x;
  const float* cb = coef + (size_t)iter * NBANDS * SIZE_N;
  for (int k = threadIdx.x; k < SIZE_N; k += 256) {
    float v = 0.f;
    if (k == j) v = cb[j];
#pragma unroll
    for (int f = 0; f < M_FACTORS; ++f) {
      const int d = 1 << (M_FACTORS - 1 - f);
      if (k == j - d) v = cb[(1 + f)  * SIZE_N + j];
      if (k == j + d) v = cb[(11 + f) * SIZE_N + j];
    }
    P[(size_t)j * SIZE_N + k] = v;
  }
}

// ---------------------------------------------------------------------------
// 3) Pout = M_iter * Pin  (banded x dense: 21 scaled row-FMAs per row).
// ---------------------------------------------------------------------------
__device__ __forceinline__ float4 compose_row(const float* __restrict__ coef,
                                              const float* __restrict__ Pin,
                                              int iter, int j, int t) {
  const float4* Pin4 = (const float4*)Pin;
  const float* cb = coef + (size_t)iter * NBANDS * SIZE_N;
  float4 acc = make_float4(0.f, 0.f, 0.f, 0.f);
#pragma unroll
  for (int b = 0; b < NBANDS; ++b) {
    int off;
    if (b == 0)       off = 0;
    else if (b <= 10) off = -(1 << (M_FACTORS - 1 - (b - 1)));
    else              off =  (1 << (M_FACTORS - 1 - (b - 11)));
    const int src = j + off;
    if (src >= 0 && src < SIZE_N) {
      const float c = cb[b * SIZE_N + j];
      const float4 p = Pin4[(size_t)src * (SIZE_N / 4) + t];
      acc.x += c * p.x; acc.y += c * p.y; acc.z += c * p.z; acc.w += c * p.w;
    }
  }
  return acc;
}

__global__ __launch_bounds__(256) void band_compose(const float* __restrict__ coef,
                                                    const float* __restrict__ Pin,
                                                    float* __restrict__ Pout, int iter) {
  const int j = blockIdx.x;
  const int t = threadIdx.x;
  ((float4*)Pout)[(size_t)j * (SIZE_N / 4) + t] = compose_row(coef, Pin, iter, j, t);
}

// Final compose (iter == 0) fused with the bf16 convert: writes T row-major.
__global__ __launch_bounds__(256) void band_compose_bf16(const float* __restrict__ coef,
                                                         const float* __restrict__ Pin,
                                                         unsigned short* __restrict__ Tb) {
  const int j = blockIdx.x;
  const int t = threadIdx.x;
  const float4 acc = compose_row(coef, Pin, 0, j, t);
  ((v4bf*)Tb)[(size_t)j * (SIZE_N / 4) + t] = cvt4(acc);
}

// ---------------------------------------------------------------------------
// 4) C[b][j] = sum_k A[b][k] * T[j][k]    (NT GEMM, both K-contiguous)
//    Block 128x64, 8 waves (4x2), 32x32 per wave as 2x2 WMMA tiles.
//    Double-buffered LDS; B tile moved with GLOBAL_LOAD_ASYNC_TO_LDS_B128
//    (ASYNCcnt) when available; A converted f32->bf16 with hardware cvt.
// ---------------------------------------------------------------------------
#define BM 128
#define BN 64
#define KC 32
#define LDA 40   // padded LDS row stride (elements); 80 B rows keep 16B chunks aligned
#define LDB 40

__global__ __launch_bounds__(256) void butterfly_gemm(
    const float* __restrict__ A, const unsigned short* __restrict__ Bt,
    float* __restrict__ C) {
  __shared__ unsigned short sA[2][BM * LDA];   // 2 x 10240 B
  __shared__ unsigned short sB[2][BN * LDB];   // 2 x  5120 B

  const int tid   = threadIdx.x;
  const int lane  = tid & 31;
  const int wave  = tid >> 5;
  const int waveM = wave & 3;
  const int waveN = wave >> 2;
  const int l16   = lane & 15;
  const int half  = lane >> 4;
  const long blockM = (long)blockIdx.x * BM;
  const long blockN = (long)blockIdx.y * BN;

  // per-thread staging coordinates
  const int arow = tid >> 1;               // A: 128 rows / 256 thr -> 2 thr per row
  const int acol = (tid & 1) << 4;         //    each thread: 16 consecutive floats
  const int brow = tid >> 2;               // B: 64 rows, 4 x16B chunks per row
  const int bcol = (tid & 3) << 3;         //    chunk start (elements)

  v8f acc[2][2] = {};

  // ---- stage one K-tile into LDS buffer `buf` ----
  auto stage = [&](int k0, int buf) {
    const float* ap = A + (blockM + arow) * (long)SIZE_N + k0 + acol;
    const float4 v0 = *(const float4*)(ap + 0);
    const float4 v1 = *(const float4*)(ap + 4);
    const float4 v2 = *(const float4*)(ap + 8);
    const float4 v3 = *(const float4*)(ap + 12);
    unsigned short* arp = &sA[buf][arow * LDA + acol];
    *(v4bf*)(arp + 0)  = cvt4(v0);
    *(v4bf*)(arp + 4)  = cvt4(v1);
    *(v4bf*)(arp + 8)  = cvt4(v2);
    *(v4bf*)(arp + 12) = cvt4(v3);
#if USE_ASYNC_LDS
    __builtin_amdgcn_global_load_async_to_lds_b128(
        (as1_v4i)(Bt + (blockN + brow) * (long)SIZE_N + k0 + bcol),
        (as3_v4i)(&sB[buf][brow * LDB + bcol]), 0, 0);
#else
    const us4* bp = (const us4*)(Bt + (blockN + brow) * (long)SIZE_N + k0 + bcol);
    us4* bdst = (us4*)(&sB[buf][brow * LDB + bcol]);
    bdst[0] = bp[0];
    bdst[1] = bp[1];
#endif
  };

  stage(0, 0);
#if USE_ASYNC_LDS
  __builtin_amdgcn_s_wait_asynccnt(0);
#endif
  __syncthreads();

  const int NT = SIZE_N / KC;   // 32
  for (int t = 0; t < NT; ++t) {
    const int cur = t & 1;
    if (t + 1 < NT) stage((t + 1) * KC, cur ^ 1);
    if (t + 2 < NT)   // warm L2 for the tile after next
      __builtin_prefetch(A + (blockM + arow) * (long)SIZE_N + (t + 2) * KC + acol, 0, 1);

    // fragments: 16-bit A/B lane layout -> two 8-elem chunks at K=8*half, +16
    v16bf af[2], bfr[2];
#pragma unroll
    for (int mi = 0; mi < 2; ++mi) {
      const int m = waveM * 32 + mi * 16 + l16;
      const v8bf lo = *(const v8bf*)(&sA[cur][m * LDA + half * 8]);
      const v8bf hi = *(const v8bf*)(&sA[cur][m * LDA + half * 8 + 16]);
      af[mi] = __builtin_shufflevector(lo, hi, 0,1,2,3,4,5,6,7,8,9,10,11,12,13,14,15);
    }
#pragma unroll
    for (int ni = 0; ni < 2; ++ni) {
      const int n = waveN * 32 + ni * 16 + l16;
      const v8bf lo = *(const v8bf*)(&sB[cur][n * LDB + half * 8]);
      const v8bf hi = *(const v8bf*)(&sB[cur][n * LDB + half * 8 + 16]);
      bfr[ni] = __builtin_shufflevector(lo, hi, 0,1,2,3,4,5,6,7,8,9,10,11,12,13,14,15);
    }
#pragma unroll
    for (int mi = 0; mi < 2; ++mi)
#pragma unroll
      for (int ni = 0; ni < 2; ++ni)
        acc[mi][ni] = __builtin_amdgcn_wmma_f32_16x16x32_bf16(
            false, af[mi], false, bfr[ni], (short)0, acc[mi][ni], false, false);

#if USE_ASYNC_LDS
    __builtin_amdgcn_s_wait_asynccnt(0);   // next-tile B landed in LDS
#endif
    __syncthreads();                        // also fences the ds stores
  }

  // write back per C/D layout: VGPR v -> row (half*8 + v), col l16
#pragma unroll
  for (int mi = 0; mi < 2; ++mi) {
#pragma unroll
    for (int ni = 0; ni < 2; ++ni) {
      const long row0 = blockM + waveM * 32 + mi * 16 + half * 8;
      const long col  = blockN + waveN * 32 + ni * 16 + l16;
      float* p = C + row0 * SIZE_N + col;
#pragma unroll
      for (int v = 0; v < 8; ++v) p[(long)v * SIZE_N] = acc[mi][ni][v];
    }
  }
}

// ---------------------------------------------------------------------------
// Host launcher. Workspace (~11.2 MB of d_ws):
//   coef : 10*21*1024 f32 | P0 : 1M f32 | P1 : 1M f32 | Tb : 1M bf16
// ---------------------------------------------------------------------------
extern "C" void kernel_launch(void* const* d_in, const int* in_sizes, int n_in,
                              void* d_out, int out_size, void* d_ws, size_t ws_size,
                              hipStream_t stream) {
  const float* input = (const float*)d_in[0];
  const float* diags = (const float*)d_in[1];
  const float* sub   = (const float*)d_in[2];
  const float* sup   = (const float*)d_in[3];
  const float* logit = (const float*)d_in[4];
  float* out = (float*)d_out;

  float* coef = (float*)d_ws;
  float* P0 = coef + (size_t)N_TERMS_C * NBANDS * SIZE_N;
  float* P1 = P0 + (size_t)SIZE_N * SIZE_N;
  unsigned short* Tb = (unsigned short*)(P1 + (size_t)SIZE_N * SIZE_N);

  prep_bands<<<N_TERMS_C, 256, 0, stream>>>(logit, diags, sub, sup, coef);
  band_init<<<SIZE_N, 256, 0, stream>>>(coef, P0, N_TERMS_C - 1);

  float* pin = P0;
  float* pout = P1;
  for (int i = N_TERMS_C - 2; i >= 1; --i) {      // 8 composes: pin ends at P0
    band_compose<<<SIZE_N, 256, 0, stream>>>(coef, pin, pout, i);
    float* tswap = pin; pin = pout; pout = tswap;
  }
  band_compose_bf16<<<SIZE_N, 256, 0, stream>>>(coef, pin, Tb);  // iter 0 + cvt

  butterfly_gemm<<<dim3(BATCH_C / BM, SIZE_N / BN), 256, 0, stream>>>(input, Tb, out);
}
